// FGCtrlDecoder_13168369729761
// MI455X (gfx1250) — matrix operand order repaired and checked by reading, not compiled, not run on hardware
//
#include <hip/hip_runtime.h>
#include <hip/hip_bf16.h>
#include <math.h>

typedef __attribute__((ext_vector_type(16))) __bf16 v16bf;
typedef __attribute__((ext_vector_type(8)))  float  v8f;

union ABFrag {
  v16bf v;
  uint4 q[2];
};

__device__ __forceinline__ unsigned short f2bf(float f) {
  unsigned int u = __float_as_uint(f);
  unsigned int r = (u + 0x7FFFu + ((u >> 16) & 1u)) >> 16;  // RNE
  return (unsigned short)r;
}
__device__ __forceinline__ float bf2f(unsigned short h) {
  return __uint_as_float(((unsigned int)h) << 16);
}

#define C_DIM 256
#define N1_DIM 512
#define YB_S 264   // bf16 row stride for LN1 output (pad 8 elems = 16B)
#define HB_S 520   // bf16 row stride for hidden (pad 8 elems = 16B)

// ---------------------------------------------------------------------------
// Prep: f32 -> bf16 weight conversion (W1: 256x512, W2: 512x256)
// ---------------------------------------------------------------------------
__global__ void cvt_weights_kernel(const float* __restrict__ W1,
                                   const float* __restrict__ W2,
                                   unsigned short* __restrict__ W1bf,
                                   unsigned short* __restrict__ W2bf) {
  int i = blockIdx.x * 256 + threadIdx.x;
  if (i < 256 * 512) {
    W1bf[i] = f2bf(W1[i]);
    W2bf[i] = f2bf(W2[i]);
  }
}

// ---------------------------------------------------------------------------
// Prep: per-(b,p) patch attention bias (K=1 attention collapses to this)
// pb[b,p,ch] = (clip[b,p,:] @ Wv_p[p] + bv_p[p]) @ Wo_p[p] + bo_p[p]
// ---------------------------------------------------------------------------
__global__ __launch_bounds__(256) void patch_bias_kernel(
    const float* __restrict__ clip, const float* __restrict__ Wv_p,
    const float* __restrict__ bv_p, const float* __restrict__ Wo_p,
    const float* __restrict__ bo_p, float* __restrict__ pb) {
  __shared__ float tmp[64];
  const int bp = blockIdx.x;        // 0..255
  const int b = bp >> 4;
  const int p = bp & 15;
  const float* cv = clip + ((size_t)b * 17 + p) * 256;
  if (threadIdx.x < 64) {
    const int d = threadIdx.x;
    const float* wv = Wv_p + (size_t)p * 256 * 64 + d;  // [p][c][d]
    float s = bv_p[p * 64 + d];
    for (int c = 0; c < 256; ++c) s += cv[c] * wv[(size_t)c * 64];
    tmp[d] = s;
  }
  __syncthreads();
  const int ch = threadIdx.x;
  const float* wo = Wo_p + (size_t)p * 64 * 256 + ch;   // [p][d][ch]
  float s = bo_p[p * 256 + ch];
  for (int d = 0; d < 64; ++d) s += tmp[d] * wo[(size_t)d * 256];
  pb[(size_t)bp * 256 + ch] = s;
}

// ---------------------------------------------------------------------------
// Prep: per-batch attention bias from the last clip token
// ab[b,ch] = (clip[b,16,:] @ Wv_a + bv_a) @ Wo_a + bo_a
// ---------------------------------------------------------------------------
__global__ __launch_bounds__(256) void attn_bias_kernel(
    const float* __restrict__ clip, const float* __restrict__ Wv_a,
    const float* __restrict__ bv_a, const float* __restrict__ Wo_a,
    const float* __restrict__ bo_a, float* __restrict__ ab) {
  __shared__ float tmp[128];
  const int b = blockIdx.x;
  const float* icv = clip + ((size_t)b * 17 + 16) * 256;
  if (threadIdx.x < 128) {
    const int d = threadIdx.x;
    float s = bv_a[d];
    for (int c = 0; c < 256; ++c) s += icv[c] * Wv_a[(size_t)c * 128 + d];
    tmp[d] = s;
  }
  __syncthreads();
  const int ch = threadIdx.x;
  float s = bo_a[ch];
  for (int d = 0; d < 128; ++d) s += tmp[d] * Wo_a[(size_t)d * 256 + ch];
  ab[b * 256 + ch] = s;
}

// ---------------------------------------------------------------------------
// Fused: gather+bias -> LN1 -> GEMM1(bf16 WMMA)+gelu -> GEMM2(bf16 WMMA)
//        -> +b2 + residual -> LN2 -> transposed store.
// One block = 64 consecutive tokens (same batch, same image row).
// ---------------------------------------------------------------------------
__global__ __launch_bounds__(256) void fused_mlp_kernel(
    const float* __restrict__ img, const float* __restrict__ pb,
    const float* __restrict__ ab, const unsigned short* __restrict__ W1bf,
    const unsigned short* __restrict__ W2bf, const float* __restrict__ b1v,
    const float* __restrict__ b2v, const float* __restrict__ g1,
    const float* __restrict__ be1, const float* __restrict__ g2,
    const float* __restrict__ be2, float* __restrict__ outp) {
  __shared__ __align__(16) unsigned short ybf[64 * YB_S];  // LN1 out, bf16
  __shared__ __align__(16) unsigned short hb[64 * HB_S];   // hidden bf16 / f32 scratch
  __shared__ float redS[64 * 4];
  __shared__ float redQ[64 * 4];
  __shared__ float mean_s[64];
  __shared__ float rstd_s[64];

  float* xF = (float*)hb;  // [64][256] f32 view: x staging, then out_pre

  const int tid = threadIdx.x;
  const int t0 = blockIdx.x * 64;
  const int b = t0 >> 12;             // 4096 tokens per batch
  const int h_idx = (t0 & 4095) >> 6; // whole tile shares one image row
  const int m_of = tid & 63;          // token in tile == w index
  const int c0 = tid >> 6;            // 0..3 channel phase

  // ---- gather x = image(transposed) + patch_bias + attn_bias ----
  const float* imgb = img + (((size_t)b * C_DIM) << 12) + (h_idx << 6);
  const float* pbb =
      pb + (((size_t)b * 16 + ((h_idx >> 4) << 2) + (m_of >> 4)) << 8);
  const float* abb = ab + (b << 8);
  for (int ch = c0; ch < C_DIM; ch += 4) {
    xF[m_of * C_DIM + ch] =
        imgb[((size_t)ch << 12) + m_of] + pbb[ch] + abb[ch];
  }
  __syncthreads();

  // ---- LN1 ----
  {
    float s = 0.f, q = 0.f;
    for (int ch = c0; ch < C_DIM; ch += 4) {
      float v = xF[m_of * C_DIM + ch];
      s += v;
      q += v * v;
    }
    redS[m_of * 4 + c0] = s;
    redQ[m_of * 4 + c0] = q;
  }
  __syncthreads();
  if (tid < 64) {
    float s = redS[tid * 4] + redS[tid * 4 + 1] + redS[tid * 4 + 2] + redS[tid * 4 + 3];
    float q = redQ[tid * 4] + redQ[tid * 4 + 1] + redQ[tid * 4 + 2] + redQ[tid * 4 + 3];
    float m = s * (1.0f / 256.0f);
    float var = q * (1.0f / 256.0f) - m * m;
    mean_s[tid] = m;
    rstd_s[tid] = rsqrtf(var + 1e-5f);
  }
  __syncthreads();
  {
    float m = mean_s[m_of], r = rstd_s[m_of];
    for (int ch = c0; ch < C_DIM; ch += 4) {
      float y = (xF[m_of * C_DIM + ch] - m) * r * g1[ch] + be1[ch];
      ybf[m_of * YB_S + ch] = f2bf(y);
    }
  }
  __syncthreads();

  const int wv = tid >> 5;           // wave 0..7 (wave32)
  const int lane = tid & 31;
  const int l15 = lane & 15;
  const int hi = (lane < 16) ? 0 : 1;
  const int akb = hi << 3;           // A-fragment K sub-offset

  // ---- GEMM1: y(64x256) @ W1(256x512), gelu, -> hb bf16 ----
  {
    const int mt = (wv & 3) << 4;    // 16-row tile
    const int nh = (wv >> 2) << 8;   // 256-col half
    ABFrag af[8];
    const int arow = mt + l15;
#pragma unroll
    for (int kt = 0; kt < 8; ++kt) {
      const uint4* pA = (const uint4*)&ybf[arow * YB_S + kt * 32 + akb];
      af[kt].q[0] = pA[0];   // K = base .. base+7
      af[kt].q[1] = pA[2];   // K = base+16 .. base+23
    }
#pragma unroll 1
    for (int nt = 0; nt < 16; ++nt) {
      const int n0 = nh + (nt << 4);
      v8f acc;
#pragma unroll
      for (int j = 0; j < 8; ++j) acc[j] = 0.f;
#pragma unroll
      for (int kt = 0; kt < 8; ++kt) {
        ABFrag bf_;  // lane = N column, 16 bf16 of K in table order
        unsigned short* bs = (unsigned short*)&bf_;
#pragma unroll
        for (int s = 0; s < 16; ++s)
          bs[s] = W1bf[(size_t)(kt * 32 + akb + (s & 7) + ((s >> 3) << 4)) * N1_DIM + n0 + l15];
        acc = __builtin_amdgcn_wmma_f32_16x16x32_bf16(
            false, af[kt].v, false, bf_.v, (short)0, acc, false, false);
      }
      const int nn = n0 + l15;
      const float bias = b1v[nn];
      const int mbase = mt + (hi << 3);
#pragma unroll
      for (int j = 0; j < 8; ++j) {
        float v = acc[j] + bias;
        float g = 0.5f * v * (1.0f + erff(v * 0.70710678118f));  // exact gelu
        hb[(mbase + j) * HB_S + nn] = f2bf(g);
      }
    }
  }
  __syncthreads();

  // ---- GEMM2: h(64x512) @ W2(512x256); accs stay in regs across barrier ----
  v8f acc2[8];
#pragma unroll
  for (int nt = 0; nt < 8; ++nt)
#pragma unroll
    for (int j = 0; j < 8; ++j) acc2[nt][j] = 0.f;

  const int mt2 = (wv & 3) << 4;
  const int nq = (wv >> 2) << 7;     // 128-col half
  const int arow2 = mt2 + l15;
#pragma unroll 1
  for (int kt = 0; kt < 16; ++kt) {
    ABFrag a2;
    const uint4* pA = (const uint4*)&hb[arow2 * HB_S + kt * 32 + akb];
    a2.q[0] = pA[0];
    a2.q[1] = pA[2];
#pragma unroll
    for (int nt = 0; nt < 8; ++nt) {
      ABFrag b2f;
      unsigned short* bs2 = (unsigned short*)&b2f;
#pragma unroll
      for (int s = 0; s < 16; ++s)
        bs2[s] = W2bf[(size_t)(kt * 32 + akb + (s & 7) + ((s >> 3) << 4)) * C_DIM + nq + (nt << 4) + l15];
      acc2[nt] = __builtin_amdgcn_wmma_f32_16x16x32_bf16(
          false, a2.v, false, b2f.v, (short)0, acc2[nt], false, false);
    }
  }
  __syncthreads();  // all hb reads done; safe to reuse region as f32 out_pre

  {
    const int mbase = mt2 + (hi << 3);
#pragma unroll
    for (int nt = 0; nt < 8; ++nt) {
      const int nn = nq + (nt << 4) + l15;
      const float bias = b2v[nn];
#pragma unroll
      for (int j = 0; j < 8; ++j) {
        const int mr = mbase + j;
        float y = bf2f(ybf[mr * YB_S + nn]);  // residual (LN1 output)
        xF[mr * C_DIM + nn] = acc2[nt][j] + bias + y;
      }
    }
  }
  __syncthreads();

  // ---- LN2 + transposed coalesced store ----
  {
    float s = 0.f, q = 0.f;
    for (int ch = c0; ch < C_DIM; ch += 4) {
      float v = xF[m_of * C_DIM + ch];
      s += v;
      q += v * v;
    }
    redS[m_of * 4 + c0] = s;
    redQ[m_of * 4 + c0] = q;
  }
  __syncthreads();
  if (tid < 64) {
    float s = redS[tid * 4] + redS[tid * 4 + 1] + redS[tid * 4 + 2] + redS[tid * 4 + 3];
    float q = redQ[tid * 4] + redQ[tid * 4 + 1] + redQ[tid * 4 + 2] + redQ[tid * 4 + 3];
    float m = s * (1.0f / 256.0f);
    float var = q * (1.0f / 256.0f) - m * m;
    mean_s[tid] = m;
    rstd_s[tid] = rsqrtf(var + 1e-5f);
  }
  __syncthreads();
  {
    float m = mean_s[m_of], r = rstd_s[m_of];
    float* ob = outp + (((size_t)b * C_DIM) << 12) + (h_idx << 6);
    for (int ch = c0; ch < C_DIM; ch += 4) {
      float v = (xF[m_of * C_DIM + ch] - m) * r * g2[ch] + be2[ch];
      ob[((size_t)ch << 12) + m_of] = v;  // out[b,ch,h,w]
    }
  }
}

// ---------------------------------------------------------------------------
extern "C" void kernel_launch(void* const* d_in, const int* in_sizes, int n_in,
                              void* d_out, int out_size, void* d_ws,
                              size_t ws_size, hipStream_t stream) {
  (void)in_sizes; (void)n_in; (void)out_size; (void)ws_size;
  const float* clip  = (const float*)d_in[0];
  const float* image = (const float*)d_in[1];
  // d_in[2] pos_embedding: dead (only feeds k of a K=1 attention)
  // d_in[3..6] Wq_p/bq_p/Wk_p/bk_p: dead (softmax over K=1 is constant 1)
  const float* Wv_p = (const float*)d_in[7];
  const float* bv_p = (const float*)d_in[8];
  const float* Wo_p = (const float*)d_in[9];
  const float* bo_p = (const float*)d_in[10];
  // d_in[11..14] Wq_a/bq_a/Wk_a/bk_a: dead
  const float* Wv_a = (const float*)d_in[15];
  const float* bv_a = (const float*)d_in[16];
  const float* Wo_a = (const float*)d_in[17];
  const float* bo_a = (const float*)d_in[18];
  const float* ln1_g = (const float*)d_in[19];
  const float* ln1_b = (const float*)d_in[20];
  const float* ln2_g = (const float*)d_in[21];
  const float* ln2_b = (const float*)d_in[22];
  const float* W1 = (const float*)d_in[23];
  const float* b1 = (const float*)d_in[24];
  const float* W2 = (const float*)d_in[25];
  const float* b2 = (const float*)d_in[26];

  // Workspace layout (~784 KB)
  unsigned short* W1bf = (unsigned short*)d_ws;        // 256*512 bf16
  unsigned short* W2bf = W1bf + 256 * 512;             // 512*256 bf16
  float* pb = (float*)(W2bf + 512 * 256);              // 16*16*256 f32
  float* ab = pb + 16 * 16 * 256;                      // 16*256 f32

  cvt_weights_kernel<<<512, 256, 0, stream>>>(W1, W2, W1bf, W2bf);
  patch_bias_kernel<<<256, 256, 0, stream>>>(clip, Wv_p, bv_p, Wo_p, bo_p, pb);
  attn_bias_kernel<<<16, 256, 0, stream>>>(clip, Wv_a, bv_a, Wo_a, bo_a, ab);
  fused_mlp_kernel<<<1024, 256, 0, stream>>>(image, pb, ab, W1bf, W2bf, b1, b2,
                                             ln1_g, ln1_b, ln2_g, ln2_b,
                                             (float*)d_out);
}
